// PKRNN_1CM_36146444763936
// MI455X (gfx1250) — matrix operand order repaired and checked by reading, not compiled, not run on hardware
//
#include <hip/hip_runtime.h>
#include <hip/hip_bf16.h>
#include <math.h>

// Problem dims
#define B_    256
#define T_    256
#define H_    1024
#define HP_   1032      // LDS row pitch for h (8 bf16 pad -> bank-conflict-free)
#define G3_   3072      // 3*H
#define INP_  131       // E + NC + 1
#define KX_   160       // gi K padded to multiple of 32 (130 -> 160)
#define E_    128
#define NCODE_ 16

typedef __attribute__((ext_vector_type(8)))  __bf16 v8bf;
typedef __attribute__((ext_vector_type(16))) __bf16 v16bf;
typedef __attribute__((ext_vector_type(8)))  float  v8f;

__device__ inline __bf16 f2bf(float f) {
  unsigned u = __builtin_bit_cast(unsigned, f);
  unsigned r = u + 0x7FFFu + ((u >> 16) & 1u);   // round-to-nearest-even
  unsigned short s = (unsigned short)(r >> 16);
  return __builtin_bit_cast(__bf16, s);
}
__device__ inline __bf16 bf_zero() {
  return __builtin_bit_cast(__bf16, (unsigned short)0);
}
__device__ inline v16bf cat16(v8bf lo, v8bf hi) {
  return __builtin_shufflevector(lo, hi, 0,1,2,3,4,5,6,7,8,9,10,11,12,13,14,15);
}

// ---------------------------------------------------------------------------
// Prep kernels
// ---------------------------------------------------------------------------
__global__ __launch_bounds__(256)
void pkrnn_prep_whh(const float* __restrict__ Whh, __bf16* __restrict__ Whhb) {
  size_t i = (size_t)blockIdx.x * 256 + threadIdx.x;
  if (i < (size_t)G3_ * H_) Whhb[i] = f2bf(Whh[i]);
}

__global__ __launch_bounds__(256)
void pkrnn_prep_wih(const float* __restrict__ Wih, __bf16* __restrict__ W1b,
                    float* __restrict__ wcol) {
  int i = blockIdx.x * 256 + threadIdx.x;       // r*KX_ + j
  if (i >= G3_ * KX_) return;
  int r = i / KX_;
  int j = i - r * KX_;
  W1b[i] = (j < 130) ? f2bf(Wih[(size_t)r * INP_ + j]) : bf_zero();
  if (j == 0) wcol[r] = Wih[(size_t)r * INP_ + 130];   // fb column
}

// biasRZ[j<2048] = bih[j]+bhh[j];  biasNI[j]=bih[2048+j];  biasNH[j]=bhh[2048+j]
__global__ __launch_bounds__(256)
void pkrnn_prep_bias(const float* __restrict__ bih, const float* __restrict__ bhh,
                     float* __restrict__ biasRZ, float* __restrict__ biasNI,
                     float* __restrict__ biasNH) {
  int j = blockIdx.x * 256 + threadIdx.x;
  if (j >= G3_) return;
  if (j < 2048) biasRZ[j] = bih[j] + bhh[j];
  else { biasNI[j - 2048] = bih[j]; biasNH[j - 2048] = bhh[j]; }
}

// Build X: (T, B, 160) bf16 = [embed-sum(128), cont(2), zero-pad(30)]
__global__ __launch_bounds__(128)
void pkrnn_embed(const float* __restrict__ embed, const float* __restrict__ cont,
                 const int* __restrict__ cat, __bf16* __restrict__ Xt) {
  int bt = blockIdx.x;                 // b*T + t
  int b = bt >> 8, t = bt & 255;
  int f = threadIdx.x;                 // 0..127
  const int* idx = cat + (size_t)bt * NCODE_;
  float s = 0.f;
  #pragma unroll
  for (int c = 0; c < NCODE_; ++c) s += embed[(size_t)idx[c] * E_ + f];
  __bf16* xr = Xt + ((size_t)t * B_ + b) * KX_;
  xr[f] = f2bf(s);
  if (f < 2)   xr[E_ + f] = f2bf(cont[(size_t)bt * 2 + f]);
  if (f >= 98) xr[130 + (f - 98)] = bf_zero();   // pad 130..159
}

// ---------------------------------------------------------------------------
// Persistent scan: 16 WGs x 512 threads; WG w owns samples [16w, 16w+16).
// No cross-WG dependency exists (recurrence is per-sample).
// Tile ownership per wave (static): r/z tiles [8w, 8w+8), n tiles [128+4w, 128+4w+4).
// ---------------------------------------------------------------------------
__global__ __launch_bounds__(512)
void pkrnn_scan(const __bf16* __restrict__ Xt,   const __bf16* __restrict__ W1b,
                const __bf16* __restrict__ Whhb, const float* __restrict__ wcol,
                const float* __restrict__ biasRZ, const float* __restrict__ biasNI,
                const float* __restrict__ biasNH,
                const float* __restrict__ Wout,  const float* __restrict__ bout,
                const float* __restrict__ dose,  const float* __restrict__ td,
                const float* __restrict__ lb,    const int* __restrict__ maskp,
                float* __restrict__ Grz, float* __restrict__ Gni,
                float* __restrict__ Gnh, float* __restrict__ hf,
                float* __restrict__ out, float* __restrict__ Kout,
                float* __restrict__ Vout) {
  __shared__ __bf16 hbL[16][HP_];      // h in bf16 (padded pitch), GEMM A-side
  __shared__ float fbL[16], tmL[16];

  const int tid  = threadIdx.x;
  const int wg   = blockIdx.x;         // 0..15
  const int m0   = wg * 16;            // first batch row of this WG
  const int lane = tid & 31;
  const int wave = tid >> 5;           // 0..15
  const int lrow = lane & 15;          // A row within tile / B col within tile
  const bool hiHalf = lane >= 16;

  // --- init state to zero ---
  for (int i = tid; i < 16 * HP_; i += 512) hbL[i / HP_][i % HP_] = bf_zero();
  for (int i = tid; i < 16 * H_; i += 512)
    hf[(size_t)(m0 + (i >> 10)) * H_ + (i & (H_ - 1))] = 0.f;
  if (tid < 16) { fbL[tid] = 0.f; tmL[tid] = 0.f; }
  __syncthreads();

  // Per-lane K-chunk offsets per the CDNA5 WMMA VGPR layout tables.
  const int aoff0 = hiHalf ? 8 : 0;        // A: {kk+0..7 | kk+8..15}
  const int aoff1 = 16 + aoff0;            // A: {kk+16..23 | kk+24..31}
  const int boff  = hiHalf ? 8 : 0;        // B: same k order as A
  const int rhalf = hiHalf ? 8 : 0;        // C/D: M = i + rhalf

  // Lane-adjusted LDS base for A-side of gh (all further offsets are immediates)
  const __bf16* hA = &hbL[lrow][aoff0];

  for (int t = 0; t < T_; ++t) {
    const __bf16* XA = Xt + ((size_t)t * B_ + (m0 + lrow)) * KX_ + aoff0;
    float fbr[8];
    #pragma unroll
    for (int i = 0; i < 8; ++i) fbr[i] = fbL[rhalf + i];

    // =================== r/z groups: 2 groups of 4 ntiles ===================
    for (int g = 0; g < 2; ++g) {
      const int colbase = (wave * 8 + g * 4) * 16 + lrow;   // < 2048
      const __bf16* W1B = W1b  + (size_t)colbase * KX_ + boff;
      const __bf16* WhB = Whhb + (size_t)colbase * H_  + boff;

      v8f acc[4];
      {
        v16bf aa[5];
        #pragma unroll
        for (int kc = 0; kc < 5; ++kc)
          aa[kc] = cat16(*(const v8bf*)(XA + kc * 32),
                         *(const v8bf*)(XA + kc * 32 + 16));
        #pragma unroll
        for (int u = 0; u < 4; ++u) {
          v16bf bb[5];
          #pragma unroll
          for (int kc = 0; kc < 5; ++kc)
            bb[kc] = cat16(*(const v8bf*)(W1B + u * 16 * KX_ + kc * 32),
                           *(const v8bf*)(W1B + u * 16 * KX_ + kc * 32 + 16));
          const float wc = wcol[colbase + u * 16];
          const float bz = biasRZ[colbase + u * 16];
          v8f a0;
          #pragma unroll
          for (int i = 0; i < 8; ++i) a0[i] = bz + fbr[i] * wc;
          #pragma unroll
          for (int kc = 0; kc < 5; ++kc)
            a0 = __builtin_amdgcn_wmma_f32_16x16x32_bf16(false, aa[kc], false, bb[kc],
                                                         (short)0, a0, false, false);
          acc[u] = a0;
        }
      }
      // gh: K outer, LDS A-chunk loaded once per 4 ntiles, imm-offset loads
      for (int kc = 0; kc < 32; kc += 2) {
        const int kk0 = kc * 32, kk1 = kk0 + 32;
        v16bf a0 = cat16(*(const v8bf*)(hA + kk0),
                         *(const v8bf*)(hA + kk0 + 16));
        v16bf a1 = cat16(*(const v8bf*)(hA + kk1),
                         *(const v8bf*)(hA + kk1 + 16));
        v16bf b0[4], b1[4];
        #pragma unroll
        for (int u = 0; u < 4; ++u) {
          b0[u] = cat16(*(const v8bf*)(WhB + u * 16 * H_ + kk0),
                        *(const v8bf*)(WhB + u * 16 * H_ + kk0 + 16));
          b1[u] = cat16(*(const v8bf*)(WhB + u * 16 * H_ + kk1),
                        *(const v8bf*)(WhB + u * 16 * H_ + kk1 + 16));
        }
        #pragma unroll
        for (int u = 0; u < 4; ++u)
          acc[u] = __builtin_amdgcn_wmma_f32_16x16x32_bf16(false, a0, false, b0[u],
                                                           (short)0, acc[u], false, false);
        #pragma unroll
        for (int u = 0; u < 4; ++u)
          acc[u] = __builtin_amdgcn_wmma_f32_16x16x32_bf16(false, a1, false, b1[u],
                                                           (short)0, acc[u], false, false);
      }
      float* grzB = Grz + (size_t)(m0 + rhalf) * 2048 + colbase;
      #pragma unroll
      for (int u = 0; u < 4; ++u)
        #pragma unroll
        for (int i = 0; i < 8; ++i)
          grzB[(size_t)i * 2048 + u * 16] = acc[u][i];
    }

    // =================== n group: 4 ntiles (gi and gh separate) =============
    {
      const int colbase = (128 + wave * 4) * 16 + lrow;     // in [2048, 3072)
      const int c2 = colbase - 2048;
      const __bf16* W1B = W1b  + (size_t)colbase * KX_ + boff;
      const __bf16* WhB = Whhb + (size_t)colbase * H_  + boff;
      float* gniB = Gni + (size_t)(m0 + rhalf) * H_ + c2;
      float* gnhB = Gnh + (size_t)(m0 + rhalf) * H_ + c2;

      v8f acc[4];
      {
        v16bf aa[5];
        #pragma unroll
        for (int kc = 0; kc < 5; ++kc)
          aa[kc] = cat16(*(const v8bf*)(XA + kc * 32),
                         *(const v8bf*)(XA + kc * 32 + 16));
        #pragma unroll
        for (int u = 0; u < 4; ++u) {
          v16bf bb[5];
          #pragma unroll
          for (int kc = 0; kc < 5; ++kc)
            bb[kc] = cat16(*(const v8bf*)(W1B + u * 16 * KX_ + kc * 32),
                           *(const v8bf*)(W1B + u * 16 * KX_ + kc * 32 + 16));
          const float wc  = wcol[colbase + u * 16];
          const float bni = biasNI[c2 + u * 16];
          v8f a0;
          #pragma unroll
          for (int i = 0; i < 8; ++i) a0[i] = bni + fbr[i] * wc;
          #pragma unroll
          for (int kc = 0; kc < 5; ++kc)
            a0 = __builtin_amdgcn_wmma_f32_16x16x32_bf16(false, aa[kc], false, bb[kc],
                                                         (short)0, a0, false, false);
          // gi part of n-gate is final -> store now
          #pragma unroll
          for (int i = 0; i < 8; ++i) gniB[(size_t)i * H_ + u * 16] = a0[i];
          const float bnh = biasNH[c2 + u * 16];
          #pragma unroll
          for (int i = 0; i < 8; ++i) acc[u][i] = bnh;   // seed gh chain
        }
      }
      for (int kc = 0; kc < 32; kc += 2) {
        const int kk0 = kc * 32, kk1 = kk0 + 32;
        v16bf a0 = cat16(*(const v8bf*)(hA + kk0),
                         *(const v8bf*)(hA + kk0 + 16));
        v16bf a1 = cat16(*(const v8bf*)(hA + kk1),
                         *(const v8bf*)(hA + kk1 + 16));
        v16bf b0[4], b1[4];
        #pragma unroll
        for (int u = 0; u < 4; ++u) {
          b0[u] = cat16(*(const v8bf*)(WhB + u * 16 * H_ + kk0),
                        *(const v8bf*)(WhB + u * 16 * H_ + kk0 + 16));
          b1[u] = cat16(*(const v8bf*)(WhB + u * 16 * H_ + kk1),
                        *(const v8bf*)(WhB + u * 16 * H_ + kk1 + 16));
        }
        #pragma unroll
        for (int u = 0; u < 4; ++u)
          acc[u] = __builtin_amdgcn_wmma_f32_16x16x32_bf16(false, a0, false, b0[u],
                                                           (short)0, acc[u], false, false);
        #pragma unroll
        for (int u = 0; u < 4; ++u)
          acc[u] = __builtin_amdgcn_wmma_f32_16x16x32_bf16(false, a1, false, b1[u],
                                                           (short)0, acc[u], false, false);
      }
      #pragma unroll
      for (int u = 0; u < 4; ++u)
        #pragma unroll
        for (int i = 0; i < 8; ++i)
          gnhB[(size_t)i * H_ + u * 16] = acc[u][i];
    }
    __syncthreads();

    // ============ Elementwise GRU + head + PK: one wave per sample ==========
    {
      const int bl = wave;               // 0..15
      const int b  = m0 + bl;
      float dk = 0.f, dv = 0.f;
      const float* grzRow = Grz + (size_t)b * 2048;
      const float* gniRow = Gni + (size_t)b * H_;
      const float* gnhRow = Gnh + (size_t)b * H_;
      float* hfRow = hf + (size_t)b * H_;
      for (int jj = 0; jj < 32; ++jj) {
        const int j = lane + (jj << 5);
        // Biases and fb*wcol are already folded into the G buffers.
        float r = 1.f / (1.f + expf(-grzRow[j]));
        float z = 1.f / (1.f + expf(-grzRow[1024 + j]));
        float n = tanhf(gniRow[j] + r * gnhRow[j]);
        float hp = hfRow[j];
        float hn = (1.f - z) * n + z * hp;
        hfRow[j]   = hn;
        hbL[bl][j] = f2bf(hn);
        dk += hn * Wout[j];
        dv += hn * Wout[1024 + j];
      }
      #pragma unroll
      for (int m = 16; m > 0; m >>= 1) {
        dk += __shfl_xor(dk, m, 32);
        dv += __shfl_xor(dv, m, 32);
      }
      if (lane == 0) {
        const size_t bt = (size_t)b * T_ + t;
        float k    = expf(dk + bout[0]);
        float v    = expf(dv + bout[1]);
        float td24 = td[bt] * 24.f;
        float d    = dose[bt];
        float Aa = expf(-k * td24);
        float Bc = (1.f / k) * (1.f - expf(-k * d)) * expf(-k * (td24 - d));
        float tm2 = tmL[bl] * Aa + Bc;
        float concn = tm2 / (v + 1e-6f);
        tmL[bl]  = tm2;
        out[bt]  = concn;
        Kout[bt] = k;
        Vout[bt] = v;
        float mf = (float)maskp[bt];
        fbL[bl] = concn * (1.f - mf) + lb[bt] * mf;   // fb for step t+1
      }
    }
    __syncthreads();
  }
}

// ---------------------------------------------------------------------------
// Loss reduction (single block)
// ---------------------------------------------------------------------------
__global__ __launch_bounds__(256)
void pkrnn_loss(const float* __restrict__ out, const float* __restrict__ Kout,
                const float* __restrict__ Vout, const float* __restrict__ lb,
                const float* __restrict__ vanco, const float* __restrict__ vfw,
                float* __restrict__ lossOut) {
  __shared__ float sm[8];
  const int tid = threadIdx.x;
  float s_se = 0.f, s_m = 0.f, s_dk = 0.f, s_dv = 0.f, s_k0 = 0.f, s_v0 = 0.f;
  for (int i = tid; i < B_ * T_; i += 256) {
    int t = i & (T_ - 1);
    float o = out[i], l = lb[i];
    float m = (l != 0.f) ? 1.f : 0.f;
    float d = l - o;
    s_se += d * d * m;
    s_m  += m;
    if (t > 0) {
      float dK = Kout[i] - Kout[i - 1]; s_dk += dK * dK;
      float dV = Vout[i] - Vout[i - 1]; s_dv += dV * dV;
    }
  }
  {
    const size_t bt0 = (size_t)tid * T_;   // tid == b (0..255)
    float e  = vanco[bt0] - Kout[bt0];  s_k0 = e * e;
    float e2 = vfw[bt0]   - Vout[bt0];  s_v0 = e2 * e2;
  }
  auto red = [&](float v) -> float {
    #pragma unroll
    for (int m = 16; m > 0; m >>= 1) v += __shfl_xor(v, m, 32);
    __syncthreads();
    if ((tid & 31) == 0) sm[tid >> 5] = v;
    __syncthreads();
    float r = 0.f;
    if (tid == 0) for (int w = 0; w < 8; ++w) r += sm[w];
    return r;
  };
  float t_se = red(s_se), t_m = red(s_m);
  float t_dk = red(s_dk), t_dv = red(s_dv);
  float t_k0 = red(s_k0), t_v0 = red(s_v0);
  if (tid == 0) {
    float mse   = t_se / fmaxf(t_m, 1.f);
    float denom = (float)(B_ * (T_ - 1));
    float kvch  = t_dk / denom * 10000.f + t_dv / denom * 100000.f;  // SCK, SCV
    float kvl   = t_k0 / (float)B_ * 1000.f + t_v0 / (float)B_ * 1000.f; // SK, SV
    lossOut[0] = mse + kvl + kvch;
  }
}

// ---------------------------------------------------------------------------
// Launch
// ---------------------------------------------------------------------------
extern "C" void kernel_launch(void* const* d_in, const int* in_sizes, int n_in,
                              void* d_out, int out_size, void* d_ws, size_t ws_size,
                              hipStream_t stream) {
  (void)in_sizes; (void)n_in; (void)out_size; (void)ws_size;
  const float* cont  = (const float*)d_in[0];
  const float* lb    = (const float*)d_in[1];
  const float* dose  = (const float*)d_in[2];
  const float* td    = (const float*)d_in[3];
  const float* vfw   = (const float*)d_in[4];
  const float* vcl   = (const float*)d_in[5];
  const float* embed = (const float*)d_in[6];
  const float* Wih   = (const float*)d_in[7];
  const float* Whh   = (const float*)d_in[8];
  const float* bih   = (const float*)d_in[9];
  const float* bhh   = (const float*)d_in[10];
  const float* Wout  = (const float*)d_in[11];
  const float* bout  = (const float*)d_in[12];
  const int*   cat   = (const int*)d_in[13];
  const int*   mask  = (const int*)d_in[14];

  char* ws = (char*)d_ws;
  size_t off = 0;
  auto alloc = [&](size_t bytes) -> char* {
    char* p = ws + off;
    off += (bytes + 255) & ~(size_t)255;
    return p;
  };
  __bf16* Xt     = (__bf16*)alloc((size_t)T_ * B_ * KX_ * 2);   // ~21 MB
  __bf16* W1b    = (__bf16*)alloc((size_t)G3_ * KX_ * 2);       // ~1 MB
  __bf16* Whhb   = (__bf16*)alloc((size_t)G3_ * H_ * 2);        // ~6.3 MB
  float*  wcol   = (float*) alloc((size_t)G3_ * 4);
  float*  biasRZ = (float*) alloc((size_t)2048 * 4);
  float*  biasNI = (float*) alloc((size_t)H_ * 4);
  float*  biasNH = (float*) alloc((size_t)H_ * 4);
  float*  Grz    = (float*) alloc((size_t)B_ * 2048 * 4);       // 2 MB
  float*  Gni    = (float*) alloc((size_t)B_ * H_ * 4);         // 1 MB
  float*  Gnh    = (float*) alloc((size_t)B_ * H_ * 4);         // 1 MB
  float*  hf     = (float*) alloc((size_t)B_ * H_ * 4);         // 1 MB

  float* out_p   = (float*)d_out;
  float* Kout    = out_p + B_ * T_;
  float* Vout    = out_p + 2 * B_ * T_;
  float* lossOut = out_p + 3 * B_ * T_;

  pkrnn_prep_whh<<<(G3_ * H_ + 255) / 256, 256, 0, stream>>>(Whh, Whhb);
  pkrnn_prep_wih<<<(G3_ * KX_ + 255) / 256, 256, 0, stream>>>(Wih, W1b, wcol);
  pkrnn_prep_bias<<<(G3_ + 255) / 256, 256, 0, stream>>>(bih, bhh, biasRZ, biasNI, biasNH);
  pkrnn_embed<<<B_ * T_, 128, 0, stream>>>(embed, cont, cat, Xt);
  pkrnn_scan<<<16, 512, 0, stream>>>(Xt, W1b, Whhb, wcol, biasRZ, biasNI, biasNH,
                                     Wout, bout, dose, td, lb, mask,
                                     Grz, Gni, Gnh, hf, out_p, Kout, Vout);
  pkrnn_loss<<<1, 256, 0, stream>>>(out_p, Kout, Vout, lb, vcl, vfw, lossOut);
}
